// BoundaryPredictor2_23304492548384
// MI455X (gfx1250) — matrix operand order repaired and checked by reading, not compiled, not run on hardware
//
#include <hip/hip_runtime.h>
#include <hip/hip_bf16.h>
#include <math.h>

// Problem constants (from reference)
#define BATCH 8
#define LSEQ  1500
#define DMODEL 1024
#define NHEAD 8
#define HD 128
#define NSEG 300
#define LNEPS 1e-5f
#define SCALE 0.08838834764831845f   // 128^-0.5

typedef __attribute__((ext_vector_type(16))) __bf16 v16bf;
typedef __attribute__((ext_vector_type(8)))  float  v8f;

static __device__ __forceinline__ v16bf mkfrag(uint4 lo, uint4 hi) {
    union { uint4 u[2]; v16bf v; } t;
    t.u[0] = lo; t.u[1] = hi;
    return t.v;
}

// One K-step worth of operands: A fragment + 4 B fragments (16x64 strip).
struct FragSet {
    uint4 a0, a1;
    uint4 b0l, b0h, b1l, b1h, b2l, b2h, b3l, b3h;
};

static __device__ __forceinline__ void load_set(FragSet& f, const __bf16* an, const __bf16* bn) {
    f.a0  = *(const uint4*)(an);
    f.a1  = *(const uint4*)(an + 8);
    f.b0l = *(const uint4*)(bn);
    f.b0h = *(const uint4*)(bn + 8);
    f.b1l = *(const uint4*)(bn + 512);
    f.b1h = *(const uint4*)(bn + 520);
    f.b2l = *(const uint4*)(bn + 1024);
    f.b2h = *(const uint4*)(bn + 1032);
    f.b3l = *(const uint4*)(bn + 1536);
    f.b3h = *(const uint4*)(bn + 1544);
}

static __device__ __forceinline__ void mma_set(const FragSet& f, v8f& c0, v8f& c1, v8f& c2, v8f& c3) {
    const v16bf av = mkfrag(f.a0, f.a1);
    c0 = __builtin_amdgcn_wmma_f32_16x16x32_bf16(false, av, false, mkfrag(f.b0l, f.b0h), (short)0, c0, false, false);
    c1 = __builtin_amdgcn_wmma_f32_16x16x32_bf16(false, av, false, mkfrag(f.b1l, f.b1h), (short)0, c1, false, false);
    c2 = __builtin_amdgcn_wmma_f32_16x16x32_bf16(false, av, false, mkfrag(f.b2l, f.b2h), (short)0, c2, false, false);
    c3 = __builtin_amdgcn_wmma_f32_16x16x32_bf16(false, av, false, mkfrag(f.b3l, f.b3h), (short)0, c3, false, false);
}

// ---------------------------------------------------------------------------
// Pack a (D x D) f32 weight W (used as  y = x @ W^T, i.e. B[k][n] = W[n][k])
// into the per-lane contiguous layout for the 32x16 bf16 B fragment:
//   pack[(((kt*D + n)*2 + half)*16) + j] = bf16( W[n*D + kt*32 + half*16 + j] )
// so lane (half, n-local) reads 16 contiguous bf16 = two b128 loads.
// ---------------------------------------------------------------------------
__global__ __launch_bounds__(256) void pack_weight_kernel(const float* __restrict__ W,
                                                          __bf16* __restrict__ pack) {
    int idx = blockIdx.x * 256 + threadIdx.x;             // over (D/32)*D*2*16 = 1,048,576
    if (idx >= (DMODEL / 32) * DMODEL * 32) return;
    int j    = idx & 15;
    int half = (idx >> 4) & 1;
    int rest = idx >> 5;
    int n    = rest & (DMODEL - 1);
    int kt   = rest >> 10;
    int k    = kt * 32 + half * 16 + j;
    pack[idx] = (__bf16)W[(long)n * DMODEL + k];
}

// ---------------------------------------------------------------------------
// qk[h, d] = sum_{d'} q[h*128 + d'] * Wk[(h*128 + d') * D + d]
// (folds the whole K projection into an 8xD matrix)
// ---------------------------------------------------------------------------
__global__ __launch_bounds__(256) void qk_kernel(const float* __restrict__ q,
                                                 const float* __restrict__ Wk,
                                                 float* __restrict__ qk) {
    int idx = blockIdx.x * 256 + threadIdx.x;             // H*D = 8192
    if (idx >= NHEAD * DMODEL) return;
    int h = idx / DMODEL;
    int d = idx - h * DMODEL;
    float acc = 0.f;
    for (int dp = 0; dp < HD; ++dp)
        acc += q[h * HD + dp] * Wk[(long)(h * HD + dp) * DMODEL + d];
    qk[idx] = acc;
}

// ---------------------------------------------------------------------------
// Per-batch serial scan: seg_id[t] = #boundaries strictly before t.
// Segments are contiguous token runs -> record [start, end) per (b, s),
// clipped to actual_len. start = -1 marks an empty segment.
// ---------------------------------------------------------------------------
__global__ void seg_range_kernel(const float* __restrict__ bnd,
                                 const float* __restrict__ lengths,
                                 int* __restrict__ seg_start,
                                 int* __restrict__ seg_end) {
    int b = threadIdx.x;
    if (b >= BATCH) return;
    int alen = (int)(lengths[b] * (float)LSEQ);           // astype(int32) truncation
    for (int s = 0; s < NSEG; ++s) {
        seg_start[b * NSEG + s] = -1;
        seg_end[b * NSEG + s]   = -1;
    }
    int cnt = 0;
    for (int t = 0; t < LSEQ; ++t) {
        int sid = cnt;
        if (bnd[b * LSEQ + t] > 0.5f) cnt++;
        if (t < alen && sid < NSEG) {
            if (seg_start[b * NSEG + sid] < 0) seg_start[b * NSEG + sid] = t;
            seg_end[b * NSEG + sid] = t + 1;
        }
    }
}

// ---------------------------------------------------------------------------
// LayerNorm (biased var) + bf16 cast + fused per-head score:
//   scores[b,h,l] = SCALE * sum_{d in [128h,128h+128)} hn[b,l,d] * qk[h,d]
// One block per token, 256 threads * 4 elements.
// ---------------------------------------------------------------------------
__global__ __launch_bounds__(256) void ln_score_kernel(const float* __restrict__ hidden,
                                                       const float* __restrict__ gamma,
                                                       const float* __restrict__ beta,
                                                       const float* __restrict__ qk,
                                                       __bf16* __restrict__ hn,
                                                       float* __restrict__ scores) {
    const int token = blockIdx.x;                          // b*LSEQ + l
    const int tid   = threadIdx.x;
    const int d0    = tid * 4;
    const int b     = token / LSEQ;
    const int l     = token - b * LSEQ;

    __shared__ float ssum[256];
    __shared__ float ssq[256];
    __shared__ float hs[NHEAD];

    const float4 x = ((const float4*)(hidden + (long)token * DMODEL))[tid];
    if (tid < NHEAD) hs[tid] = 0.f;
    ssum[tid] = x.x + x.y + x.z + x.w;
    ssq[tid]  = x.x * x.x + x.y * x.y + x.z * x.z + x.w * x.w;
    __syncthreads();
    for (int off = 128; off > 0; off >>= 1) {
        if (tid < off) { ssum[tid] += ssum[tid + off]; ssq[tid] += ssq[tid + off]; }
        __syncthreads();
    }
    const float mu   = ssum[0] * (1.0f / DMODEL);
    const float var  = ssq[0] * (1.0f / DMODEL) - mu * mu;
    const float rstd = rsqrtf(var + LNEPS);

    const float4 g  = ((const float4*)gamma)[tid];
    const float4 bt = ((const float4*)beta)[tid];
    const float n0 = (x.x - mu) * rstd * g.x + bt.x;
    const float n1 = (x.y - mu) * rstd * g.y + bt.y;
    const float n2 = (x.z - mu) * rstd * g.z + bt.z;
    const float n3 = (x.w - mu) * rstd * g.w + bt.w;

    union { __bf16 h[4]; uint2 u; } pk;
    pk.h[0] = (__bf16)n0; pk.h[1] = (__bf16)n1; pk.h[2] = (__bf16)n2; pk.h[3] = (__bf16)n3;
    *(uint2*)(hn + (long)token * DMODEL + d0) = pk.u;

    const int h = d0 >> 7;                                 // head of this thread's 4 elems
    const float* qkh = qk + h * DMODEL + d0;
    const float part = n0 * qkh[0] + n1 * qkh[1] + n2 * qkh[2] + n3 * qkh[3];
    atomicAdd(&hs[h], part);                               // ds_add_f32
    __syncthreads();
    if (tid < NHEAD)
        scores[((long)b * NHEAD + tid) * LSEQ + l] = hs[tid] * SCALE;
}

// ---------------------------------------------------------------------------
// bf16 WMMA GEMM: C(M x 1024) = A(M x 1024, bf16 row-major) * Bpack
// Each wave computes a 16x64 output strip (4 accumulators); the A fragment is
// loaded once per K-step and reused by 4 WMMAs. The K-loop is ping-pong
// double-buffered (kt += 2, two named fragment sets) so no register rotation
// copies are needed: each set is refilled in place one step before its next
// use, keeping ~10 loads in flight behind the matrix pipe.
// 8 waves/block -> block covers 16 rows x 512 cols; grid = (M/16, 1024/512).
// OUT_F32 = 1 -> f32 output (final result), 0 -> bf16 output (v matrix).
// ---------------------------------------------------------------------------
template <int OUT_F32>
__global__ __launch_bounds__(256) void gemm_wmma_kernel(const __bf16* __restrict__ A,
                                                        const __bf16* __restrict__ Bp,
                                                        void* __restrict__ Cout) {
    const int wave = threadIdx.x >> 5;
    const int lane = threadIdx.x & 31;
    const int half = lane >> 4;
    const int r    = lane & 15;
    const int n0   = (blockIdx.y * 8 + wave) * 64;         // 4 consecutive 16-col tiles
    const long m   = (long)blockIdx.x * 16 + r;

    const __bf16* arow = A + m * DMODEL + half * 16;       // A: +32 elems per K-step
    const __bf16* bcol = Bp + (((long)(n0 + r)) * 2 + half) * 16;  // B: +D*32 elems per K-step

    v8f c0 = {}, c1 = {}, c2 = {}, c3 = {};
    FragSet X, Y;

    // prologue: X <- kt 0
    load_set(X, arow, bcol);

    constexpr int KT = DMODEL / 32;                        // 32 K-steps
    for (int kt = 0; kt < KT - 2; kt += 2) {
        // Y <- kt+1 while X (kt) computes
        load_set(Y, arow + (kt + 1) * 32, bcol + (long)(kt + 1) * (DMODEL * 32));
        mma_set(X, c0, c1, c2, c3);
        // X <- kt+2 while Y (kt+1) computes
        load_set(X, arow + (kt + 2) * 32, bcol + (long)(kt + 2) * (DMODEL * 32));
        mma_set(Y, c0, c1, c2, c3);
    }
    // tail: X holds kt = KT-2; load KT-1 into Y, then drain both
    load_set(Y, arow + (KT - 1) * 32, bcol + (long)(KT - 1) * (DMODEL * 32));
    mma_set(X, c0, c1, c2, c3);
    mma_set(Y, c0, c1, c2, c3);

    // D layout: VGPR j -> row half*8 + j, col r  (per 16x16 tile)
    const long rowbase = (long)blockIdx.x * 16 + half * 8;
    if (OUT_F32) {
        float* Co = (float*)Cout;
#pragma unroll
        for (int j = 0; j < 8; ++j) {
            float* row = Co + (rowbase + j) * DMODEL + n0 + r;
            row[0] = c0[j]; row[16] = c1[j]; row[32] = c2[j]; row[48] = c3[j];
        }
    } else {
        __bf16* Co = (__bf16*)Cout;
#pragma unroll
        for (int j = 0; j < 8; ++j) {
            __bf16* row = Co + (rowbase + j) * DMODEL + n0 + r;
            row[0] = (__bf16)c0[j]; row[16] = (__bf16)c1[j];
            row[32] = (__bf16)c2[j]; row[48] = (__bf16)c3[j];
        }
    }
}

// ---------------------------------------------------------------------------
// Segment softmax + pooling: one block per (b, s); threads own 4 channels.
// pooled[b,s,:] = sum_t softmax(scores[b,h,seg])_t * v[b,t,:]   (per head h)
// ---------------------------------------------------------------------------
__global__ __launch_bounds__(256) void pool_kernel(const float* __restrict__ scores,
                                                   const __bf16* __restrict__ v,
                                                   const int* __restrict__ seg_start,
                                                   const int* __restrict__ seg_end,
                                                   __bf16* __restrict__ pooled) {
    const int bs  = blockIdx.x;                            // b*NSEG + s
    const int b   = bs / NSEG;
    const int tid = threadIdx.x;
    const int d0  = tid * 4;
    const int h   = d0 >> 7;
    const int st  = seg_start[bs];
    const int en  = seg_end[bs];

    __shared__ float m_s[NHEAD];
    __shared__ float inv_s[NHEAD];

    if (tid < NHEAD) {
        float m = -INFINITY, se = 0.f;
        const float* sc = scores + ((long)b * NHEAD + tid) * LSEQ;
        for (int t = st; t < en; ++t) m = fmaxf(m, sc[t]);
        for (int t = st; t < en; ++t) se += __expf(sc[t] - m);
        m_s[tid]   = m;
        inv_s[tid] = (se > 0.f) ? (1.f / se) : 0.f;
    }
    __syncthreads();

    float a0 = 0.f, a1 = 0.f, a2 = 0.f, a3 = 0.f;
    if (st >= 0) {
        const float m = m_s[h], inv = inv_s[h];
        const float* sc = scores + ((long)b * NHEAD + h) * LSEQ;
        for (int t = st; t < en; ++t) {
            const float w = __expf(sc[t] - m) * inv;
            const __bf16* vp = v + ((long)b * LSEQ + t) * DMODEL + d0;
            a0 += w * (float)vp[0];
            a1 += w * (float)vp[1];
            a2 += w * (float)vp[2];
            a3 += w * (float)vp[3];
        }
    }
    union { __bf16 h4[4]; uint2 u; } pk;
    pk.h4[0] = (__bf16)a0; pk.h4[1] = (__bf16)a1; pk.h4[2] = (__bf16)a2; pk.h4[3] = (__bf16)a3;
    *(uint2*)(pooled + (long)bs * DMODEL + d0) = pk.u;
}

// ---------------------------------------------------------------------------
// Host launcher
// ---------------------------------------------------------------------------
extern "C" void kernel_launch(void* const* d_in, const int* in_sizes, int n_in,
                              void* d_out, int out_size, void* d_ws, size_t ws_size,
                              hipStream_t stream) {
    const float* hidden  = (const float*)d_in[0];   // (B, L, D)
    const float* bnd     = (const float*)d_in[1];   // (B, L)
    const float* lengths = (const float*)d_in[2];   // (B,)
    const float* lquery  = (const float*)d_in[3];   // (D,)
    const float* Wk      = (const float*)d_in[4];   // (D, D)
    const float* Wv      = (const float*)d_in[5];   // (D, D)
    const float* Wo      = (const float*)d_in[6];   // (D, D)
    const float* gamma   = (const float*)d_in[7];   // (D,)
    const float* beta    = (const float*)d_in[8];   // (D,)
    (void)in_sizes; (void)n_in; (void)out_size; (void)ws_size;

    // workspace carve-up (256B aligned)
    char* ws = (char*)d_ws;
    size_t off = 0;
    auto carve = [&](size_t bytes) { char* p = ws + off; off += (bytes + 255) & ~(size_t)255; return p; };
    int*    seg_start = (int*)   carve((size_t)BATCH * NSEG * 4);
    int*    seg_end   = (int*)   carve((size_t)BATCH * NSEG * 4);
    float*  qk        = (float*) carve((size_t)NHEAD * DMODEL * 4);
    float*  scores    = (float*) carve((size_t)BATCH * NHEAD * LSEQ * 4);
    __bf16* hn        = (__bf16*)carve((size_t)BATCH * LSEQ * DMODEL * 2);
    __bf16* vmat      = (__bf16*)carve((size_t)BATCH * LSEQ * DMODEL * 2);
    __bf16* pooled    = (__bf16*)carve((size_t)BATCH * NSEG * DMODEL * 2);
    __bf16* wv_pack   = (__bf16*)carve((size_t)DMODEL * DMODEL * 2);
    __bf16* wo_pack   = (__bf16*)carve((size_t)DMODEL * DMODEL * 2);

    const int packN = (DMODEL / 32) * DMODEL * 32;         // 1,048,576
    pack_weight_kernel<<<(packN + 255) / 256, 256, 0, stream>>>(Wv, wv_pack);
    pack_weight_kernel<<<(packN + 255) / 256, 256, 0, stream>>>(Wo, wo_pack);
    qk_kernel<<<(NHEAD * DMODEL + 255) / 256, 256, 0, stream>>>(lquery, Wk, qk);
    seg_range_kernel<<<1, 32, 0, stream>>>(bnd, lengths, seg_start, seg_end);
    ln_score_kernel<<<BATCH * LSEQ, 256, 0, stream>>>(hidden, gamma, beta, qk, hn, scores);

    // v = hn @ Wv^T : M = 12000 -> 750 row tiles; each block covers 512 cols
    gemm_wmma_kernel<0><<<dim3(750, 2), 256, 0, stream>>>(hn, wv_pack, (void*)vmat);

    pool_kernel<<<BATCH * NSEG, 256, 0, stream>>>(scores, vmat, seg_start, seg_end, pooled);

    // out = pooled @ Wo^T : M = 2400 -> 150 row tiles
    gemm_wmma_kernel<1><<<dim3(150, 2), 256, 0, stream>>>(pooled, wo_pack, d_out);
}
